// MFC_Bottleneck_29841432773493
// MI455X (gfx1250) — matrix operand order repaired and checked by reading, not compiled, not run on hardware
//
#include <hip/hip_runtime.h>
#include <hip/hip_bf16.h>

typedef __attribute__((ext_vector_type(16))) _Float16 v16h;
typedef __attribute__((ext_vector_type(8)))  _Float16 v8h;
typedef __attribute__((ext_vector_type(8)))  float    v8f;

#define Bb   8
#define Cc   128
#define Hh   128
#define Ww   128
#define NOFF 18     // 2*K*K offset channels
#define KK9  9

// ---------------------------------------------------------------------------
// Kernel W: reorder w_off [18][128][3][3] (f32) into per-lane-contiguous f16
// B-fragment order for v_wmma_f32_16x16x32_f16.
//   wBr[((step*2 + ntile)*32 + lane)*16 + j]
//   step = t*4 + kc  (tap t in 0..8, channel block kc in 0..3)
//   lane: n = ntile*16 + (lane&15); K range = (lane<16 ? 0..15 : 16..31)
//   j:    c = kc*32 + (lane>=16)*16 + j
// n >= 18 padded with zeros.
// ---------------------------------------------------------------------------
__global__ void reorder_w_kernel(const float* __restrict__ w_off,
                                 _Float16* __restrict__ wBr) {
    int e = blockIdx.x * 256 + threadIdx.x;
    const int total = 36 * 2 * 32 * 16;
    if (e >= total) return;
    int j    = e & 15;
    int lane = (e >> 4) & 31;
    int nt   = (e >> 9) & 1;
    int step = e >> 10;              // 0..35
    int t    = step >> 2;            // tap
    int kc   = step & 3;             // channel block
    int n = nt * 16 + (lane & 15);
    int c = kc * 32 + ((lane >> 4) << 4) + j;
    float v = (n < NOFF) ? w_off[(n * Cc + c) * KK9 + t] : 0.0f;
    wBr[e] = (_Float16)v;
}

// ---------------------------------------------------------------------------
// Kernel 1: offset conv as implicit GEMM with WMMA.
// Wave-per-M-tile: 16 consecutive pixels along W of one (b,h) row.
// 8192 M-tiles total -> 2048 blocks x 4 waves.
// ---------------------------------------------------------------------------
__global__ void offset_conv_wmma(const float* __restrict__ x,
                                 const _Float16* __restrict__ wBr,
                                 const float* __restrict__ b_off,
                                 float* __restrict__ off_out) {
    __shared__ __align__(16) _Float16 lds_a[4][16 * 32];

    const int lane = threadIdx.x & 31;
    const int wv   = threadIdx.x >> 5;
    const int tile = blockIdx.x * 4 + wv;        // 0..8191
    const int b    = tile >> 10;                 // 1024 tiles per image
    const int rem  = tile & 1023;
    const int h    = rem >> 3;                   // 8 tiles per row
    const int w0   = (rem & 7) << 4;

    _Float16* a = lds_a[wv];
    const int m_lane = lane & 15;
    const int hi     = lane >> 4;                // 0 or 1

    v8f acc0 = {};
    v8f acc1 = {};

    for (int t = 0; t < 9; ++t) {
        const int dy = t / 3 - 1;
        const int dx = t % 3 - 1;
        const int y  = h + dy;
        const bool yok = (y >= 0) && (y < Hh);
        for (int kc = 0; kc < 4; ++kc) {
            // ---- stage A tile: channel (kc*32 + lane), 16 pixels along W
            const int c = kc * 32 + lane;
            const float* xrow = x + (((size_t)(b * Cc + c) * Hh + y) * Ww);
            #pragma unroll
            for (int m = 0; m < 16; ++m) {
                int xc = w0 + m + dx;
                float v = 0.0f;
                if (yok && xc >= 0 && xc < Ww) v = xrow[xc];
                a[m * 32 + lane] = (_Float16)v;
            }
            asm volatile("s_wait_dscnt 0" ::: "memory");

            // ---- A fragment: contiguous k = 16*hi + j (matches wBr packing)
            v8h alo = *reinterpret_cast<const v8h*>(&a[m_lane * 32 + hi * 16]);
            v8h ahi = *reinterpret_cast<const v8h*>(&a[m_lane * 32 + hi * 16 + 8]);
            v16h afrag;
            #pragma unroll
            for (int j = 0; j < 8; ++j) { afrag[j] = alo[j]; afrag[j + 8] = ahi[j]; }

            // ---- B fragments: one contiguous 32B load each (L2-resident)
            const int step = t * 4 + kc;
            const _Float16* bp0 = wBr + ((size_t)((step * 2 + 0) * 32 + lane)) * 16;
            const _Float16* bp1 = wBr + ((size_t)((step * 2 + 1) * 32 + lane)) * 16;
            v16h bf0 = *reinterpret_cast<const v16h*>(bp0);
            v16h bf1 = *reinterpret_cast<const v16h*>(bp1);
            if (step < 35) __builtin_prefetch(bp0 + 1024, 0, 0);

            acc0 = __builtin_amdgcn_wmma_f32_16x16x32_f16(
                false, afrag, false, bf0, (short)0, acc0, false, false);
            acc1 = __builtin_amdgcn_wmma_f32_16x16x32_f16(
                false, afrag, false, bf1, (short)0, acc1, false, false);
            asm volatile("" ::: "memory");  // keep LDS reuse ordered across steps
        }
    }

    // ---- epilogue: D layout n = lane&15, m = r + 8*hi
    const int n0 = m_lane;            // 0..15, always valid (< 18)
    {
        float bias = b_off[n0];
        float* orow = off_out + (((size_t)b * NOFF + n0) * Hh + h) * Ww + w0;
        #pragma unroll
        for (int r = 0; r < 8; ++r) orow[r + hi * 8] = acc0[r] + bias;
    }
    const int n1 = 16 + m_lane;
    if (n1 < NOFF) {
        float bias = b_off[n1];
        float* orow = off_out + (((size_t)b * NOFF + n1) * Hh + h) * Ww + w0;
        #pragma unroll
        for (int r = 0; r < 8; ++r) orow[r + hi * 8] = acc1[r] + bias;
    }
}

// ---------------------------------------------------------------------------
// Kernel 2: deformable depthwise conv (bilinear gather) + BN statistics.
// One block = one (b,c,h) row of 128 pixels; all threads share channel c, so
// block-reduce sum / sumsq then one atomicAdd pair per block.
// ---------------------------------------------------------------------------
__device__ __forceinline__ float sampleZ(const float* __restrict__ p, int y, int xx) {
    if (y < 0 || y >= Hh || xx < 0 || xx >= Ww) return 0.0f;
    return p[y * Ww + xx];
}

__global__ void deform_dw_kernel(const float* __restrict__ x,
                                 const float* __restrict__ off,
                                 const float* __restrict__ w_dw,
                                 float* __restrict__ out_pre,
                                 float* __restrict__ stats) {
    const int blk = blockIdx.x;         // ((b*C + c)*H + h)
    const int w   = threadIdx.x;        // 0..127
    const int h   = blk & (Hh - 1);
    const int bc  = blk >> 7;
    const int c   = bc & (Cc - 1);
    const int b   = bc >> 7;

    const float* xb = x + (size_t)bc * Hh * Ww;                      // x[b][c]
    const float* offbase = off + ((size_t)b * NOFF) * Hh * Ww + (size_t)h * Ww + w;

    float acc = 0.0f;
    #pragma unroll
    for (int j = 0; j < 9; ++j) {
        float oy = offbase[(size_t)(2 * j)     * Hh * Ww];
        float ox = offbase[(size_t)(2 * j + 1) * Hh * Ww];
        float py = (float)h + (float)(j / 3 - 1) + oy;
        float px = (float)w + (float)(j % 3 - 1) + ox;
        float y0f = floorf(py), x0f = floorf(px);
        float ty = py - y0f, tx = px - x0f;
        int y0 = (int)y0f, x0 = (int)x0f;
        float v00 = sampleZ(xb, y0,     x0);
        float v01 = sampleZ(xb, y0,     x0 + 1);
        float v10 = sampleZ(xb, y0 + 1, x0);
        float v11 = sampleZ(xb, y0 + 1, x0 + 1);
        float val = v00 * (1.0f - ty) * (1.0f - tx) + v01 * (1.0f - ty) * tx
                  + v10 * ty * (1.0f - tx)          + v11 * ty * tx;
        acc += w_dw[c * KK9 + j] * val;
    }
    out_pre[(size_t)blk * Ww + w] = acc;

    // block reduction (sum, sumsq) -> atomics into stats[c], stats[128+c]
    __shared__ float rs[128];
    __shared__ float rq[128];
    rs[w] = acc;
    rq[w] = acc * acc;
    __syncthreads();
    for (int s = 64; s > 0; s >>= 1) {
        if (w < s) { rs[w] += rs[w + s]; rq[w] += rq[w + s]; }
        __syncthreads();
    }
    if (w == 0) {
        atomicAdd(&stats[c], rs[0]);
        atomicAdd(&stats[Cc + c], rq[0]);
    }
}

// ---------------------------------------------------------------------------
// Kernel 3: BN finalize + SiLU + residual.
// ---------------------------------------------------------------------------
__global__ void bn_silu_res_kernel(const float* __restrict__ out_pre,
                                   const float* __restrict__ x,
                                   const float* __restrict__ gamma,
                                   const float* __restrict__ beta,
                                   const float* __restrict__ stats,
                                   float* __restrict__ out) {
    size_t idx = (size_t)blockIdx.x * 256 + threadIdx.x;
    const size_t total = (size_t)Bb * Cc * Hh * Ww;
    if (idx >= total) return;
    int c = (int)((idx >> 14) & (Cc - 1));                 // H*W = 16384
    const float Ninv = 1.0f / (float)(Bb * Hh * Ww);
    float mean = stats[c] * Ninv;
    float var  = stats[Cc + c] * Ninv - mean * mean;
    float inv  = rsqrtf(var + 1e-5f);
    float v = (out_pre[idx] - mean) * inv * gamma[c] + beta[c];
    float s = v / (1.0f + __expf(-v));                     // SiLU
    out[idx] = s + x[idx];
}

// ---------------------------------------------------------------------------
extern "C" void kernel_launch(void* const* d_in, const int* in_sizes, int n_in,
                              void* d_out, int out_size, void* d_ws, size_t ws_size,
                              hipStream_t stream) {
    const float* x     = (const float*)d_in[0];
    const float* w_off = (const float*)d_in[1];
    const float* b_off = (const float*)d_in[2];
    const float* w_dw  = (const float*)d_in[3];
    const float* gamma = (const float*)d_in[4];
    const float* beta  = (const float*)d_in[5];
    float* out = (float*)d_out;

    // workspace layout (all offsets 256B-aligned)
    char* ws = (char*)d_ws;
    const size_t off_bytes = (size_t)Bb * NOFF * Hh * Ww * sizeof(float);  // 9,437,184
    const size_t pre_bytes = (size_t)Bb * Cc * Hh * Ww * sizeof(float);    // 67,108,864
    float*     off_ws  = (float*)(ws);
    float*     out_pre = (float*)(ws + off_bytes);
    _Float16*  wBr     = (_Float16*)(ws + off_bytes + pre_bytes);
    float*     stats   = (float*)(ws + off_bytes + pre_bytes + 73728);

    (void)in_sizes; (void)n_in; (void)out_size; (void)ws_size;

    // 1) reorder conv weights into WMMA B-fragment layout (f16)
    reorder_w_kernel<<<144, 256, 0, stream>>>(w_off, wBr);
    // 2) zero BN stats accumulators (ws is poisoned by harness)
    hipMemsetAsync(stats, 0, 2 * Cc * sizeof(float), stream);
    // 3) offset conv: implicit GEMM, v_wmma_f32_16x16x32_f16
    offset_conv_wmma<<<2048, 128, 0, stream>>>(x, wBr, b_off, off_ws);
    // 4) deformable depthwise sampling + BN stat accumulation
    deform_dw_kernel<<<Bb * Cc * Hh, 128, 0, stream>>>(x, off_ws, w_dw, out_pre, stats);
    // 5) BN finalize + SiLU + residual
    const size_t total = (size_t)Bb * Cc * Hh * Ww;
    bn_silu_res_kernel<<<(unsigned)((total + 255) / 256), 256, 0, stream>>>(
        out_pre, x, gamma, beta, stats, out);
}
